// MultiHeadCrossAttention_65300682768911
// MI455X (gfx1250) — matrix operand order repaired and checked by reading, not compiled, not run on hardware
//
#include <hip/hip_runtime.h>
#include <hip/hip_bf16.h>

typedef __attribute__((ext_vector_type(16))) _Float16 v16h;
typedef __attribute__((ext_vector_type(8)))  _Float16 v8h;
typedef __attribute__((ext_vector_type(8)))  float    v8f;

#define NSEQ 4096
#define BATCH 4
#define NHEAD 4
#define EMB 128
#define TOKENS (BATCH * NSEQ)   // 16384
#define KV_CHUNK 32
#define NCHUNK (NSEQ / KV_CHUNK)

// ---------------------------------------------------------------------------
// Fragment loaders (wave32; A/B k order is this code's choice, C/D per ISA)
// A-matrix 16x32 f16: lane(hh=lane>>4) holds one contiguous run K = [16h .. 16h+15] (this code's order, not ISA's)
// B-matrix 32x16 f16: lane holds one contiguous run K = [16h .. 16h+15] (this code's order, not ISA's)
// C/D 16x16 f32: VGPR r -> (M = r + 8h, N = lane&15)
// ---------------------------------------------------------------------------
__device__ inline v16h load_a32_h(const _Float16* p, int hh) {
    v16h a;
    v8h lo = *(const v8h*)(p + 16 * hh);
    v8h hi = *(const v8h*)(p + 16 * hh + 8);
#pragma unroll
    for (int i = 0; i < 8; ++i) { a[i] = lo[i]; a[8 + i] = hi[i]; }
    return a;
}

__device__ inline v16h load_a32_f(const float* p, int hh) {
    v16h a;
    v8f lo = *(const v8f*)(p + 16 * hh);
    v8f hi = *(const v8f*)(p + 16 * hh + 8);
#pragma unroll
    for (int i = 0; i < 8; ++i) { a[i] = (_Float16)lo[i]; a[8 + i] = (_Float16)hi[i]; }
    return a;
}

__device__ inline v16h load_b32_f(const float* p, int hh) {
    v16h b;
    const float* q = p + 16 * hh;
    v8f lo = *(const v8f*)(q);
    v8f hi = *(const v8f*)(q + 8);
#pragma unroll
    for (int i = 0; i < 8; ++i) { b[i] = (_Float16)lo[i]; b[8 + i] = (_Float16)hi[i]; }
    return b;
}

__device__ inline v8f wmma_f16(v16h a, v16h b, v8f c) {
    return __builtin_amdgcn_wmma_f32_16x16x32_f16(false, a, false, b, (short)0, c,
                                                  false, false);
}

// CDNA5 async Global->LDS copy (16B per lane, ASYNCcnt-tracked).
__device__ inline void async_copy16(const _Float16* g, const _Float16* lds) {
    unsigned      loff = (unsigned)(unsigned long long)lds;  // addr[31:0] = LDS offset
    unsigned long long ga = (unsigned long long)g;
    asm volatile("global_load_async_to_lds_b128 %0, %1, off"
                 :: "v"(loff), "v"(ga)
                 : "memory");
}

__device__ inline void wait_async0() {
    asm volatile("s_wait_asynccnt 0x0" ::: "memory");
}

// ---------------------------------------------------------------------------
// Kernel 1: QKV projection.  out^T = W * x^T (+ bias), 16 tokens x 128 cols
// per wave.  which = blockIdx.y selects Q/K/V.  Q is pre-scaled by Dh^-0.5.
// V is stored transposed per-head: Vt[(b*H+h)*32 + dh][n].
// ---------------------------------------------------------------------------
__global__ __launch_bounds__(256) void qkv_proj_kernel(
    const float* __restrict__ Xq, const float* __restrict__ Xk,
    const float* __restrict__ Xv, const float* __restrict__ Wq,
    const float* __restrict__ Wk, const float* __restrict__ Wv,
    const float* __restrict__ bq, const float* __restrict__ bk,
    const float* __restrict__ bv, _Float16* __restrict__ Qh,
    _Float16* __restrict__ Kh, _Float16* __restrict__ Vt) {
    const int lane = threadIdx.x & 31;
    const int wave = threadIdx.x >> 5;
    const int lq = lane & 15;
    const int hh = lane >> 4;
    const int which = blockIdx.y;

    const float* X = (which == 0) ? Xq : (which == 1) ? Xk : Xv;
    const float* W = (which == 0) ? Wq : (which == 1) ? Wk : Wv;
    const float* Bi = (which == 0) ? bq : (which == 1) ? bk : bv;
    const float postscale = (which == 0) ? 0.17677669529663687f : 1.0f;  // 32^-0.5

    const int tok = blockIdx.x * 128 + wave * 16 + lq;  // this lane's token col

    // B fragments: x^T columns = token rows of X, 4 k-chunks of 32
    const float* xr = X + (size_t)tok * EMB;
    v16h xb[4];
#pragma unroll
    for (int c = 0; c < 4; ++c) xb[c] = load_b32_f(xr + 32 * c, hh);

#pragma unroll
    for (int jt = 0; jt < 8; ++jt) {
        const int jbase = jt * 16;
        // bias into C: lane holds rows j = jbase + 8*hh + r (contiguous)
        v8f acc = *(const v8f*)(Bi + jbase + 8 * hh);
#pragma unroll
        for (int c = 0; c < 4; ++c) {
            // A fragment: W row j = jbase + lq, k-chunk c
            const float* wr = W + (size_t)(jbase + lq) * EMB + 32 * c;
            v16h wa = load_a32_f(wr, hh);
            acc = wmma_f16(wa, xb[c], acc);
        }
        if (which < 2) {
            _Float16* O = (which == 0) ? Qh : Kh;
            v8h o;
#pragma unroll
            for (int r = 0; r < 8; ++r) o[r] = (_Float16)(acc[r] * postscale);
            *(v8h*)(O + (size_t)tok * EMB + jbase + 8 * hh) = o;
        } else {
            const int b = tok >> 12;       // token / NSEQ
            const int n = tok & (NSEQ - 1);
#pragma unroll
            for (int r = 0; r < 8; ++r) {
                const int j = jbase + 8 * hh + r;
                const int head = j >> 5, dh = j & 31;
                Vt[(size_t)((b * NHEAD + head) * 32 + dh) * NSEQ + n] =
                    (_Float16)acc[r];
            }
        }
    }
}

// ---------------------------------------------------------------------------
// Kernel 2: flash attention.  One block = 8 waves = 128 query rows of one
// (b, head).  K / V^T tiles (32 kv x 32 dh, 2KB each) are staged in LDS via
// async Global->LDS copies, double buffered so the copy of chunk i+1 overlaps
// the compute of chunk i.  Each wave computes S^T = K * Q^T so softmax stats
// and the ctx^T accumulator are per-lane in the query dimension.
// ---------------------------------------------------------------------------
__global__ __launch_bounds__(256) void flash_attn_kernel(
    const _Float16* __restrict__ Qh, const _Float16* __restrict__ Kh,
    const _Float16* __restrict__ Vt, _Float16* __restrict__ Ctx) {
    __shared__ _Float16 smK[2][KV_CHUNK * 32];   // [kv_local][dh]
    __shared__ _Float16 smV[2][KV_CHUNK * 32];   // [dh][kv_local]

    const int tid = threadIdx.x;
    const int lane = tid & 31;
    const int wave = tid >> 5;
    const int lq = lane & 15;
    const int hh = lane >> 4;

    const int qblk = blockIdx.x & 31;     // 32 q-blocks of 128 rows
    const int bh = blockIdx.x >> 5;       // b*NHEAD + head
    const int b = bh >> 2, head = bh & 3;
    const int qbase = qblk * 128 + wave * 16;

    // Q^T B-fragment: column q = qbase+lq, contiguous Dh run [16h..16h+15]
    const _Float16* qrow =
        Qh + ((size_t)(b * NSEQ) + qbase + lq) * EMB + head * 32;
    const v16h qf = *(const v16h*)(qrow + 16 * hh);

    const _Float16* Kbase = Kh + (size_t)(b * NSEQ) * EMB + head * 32;
    const _Float16* Vbase = Vt + (size_t)bh * 32 * NSEQ;

    // Loader mapping: waves 0..3 copy the K tile, waves 4..7 the V^T tile.
    // 128 slots of 16B per tile: slot s -> row s>>2, 8-half part s&3.
    const int s = tid & 127;
    const int lrow = s >> 2;
    const int lpart = (s & 3) * 8;
    const bool isK = (tid < 128);
    const _Float16* gsrc0 = isK ? (Kbase + (size_t)lrow * EMB + lpart)
                                : (Vbase + (size_t)lrow * NSEQ + lpart);
    const int gstep = isK ? KV_CHUNK * EMB : KV_CHUNK;  // advance per kv chunk
    const _Float16* lds0 = isK ? &smK[0][lrow * 32 + lpart]
                               : &smV[0][lrow * 32 + lpart];
    const _Float16* lds1 = isK ? &smK[1][lrow * 32 + lpart]
                               : &smV[1][lrow * 32 + lpart];

    v8f acc0 = {}, acc1 = {};
    float m_i = -3.0e38f, l_i = 0.0f;

    async_copy16(gsrc0, lds0);  // prefetch chunk 0

    for (int i = 0; i < NCHUNK; ++i) {
        wait_async0();
        __syncthreads();        // chunk i resident; all waves done with i-1
        if (i + 1 < NCHUNK)     // prefetch chunk i+1 (overlaps compute below)
            async_copy16(gsrc0 + (size_t)(i + 1) * gstep,
                         ((i + 1) & 1) ? lds1 : lds0);

        const _Float16* kbuf = smK[i & 1];
        const _Float16* vbuf = smV[i & 1];

        // K A-fragments for kv_local 0..15 and 16..31
        v16h ka0 = load_a32_h(kbuf + lq * 32, hh);
        v16h ka1 = load_a32_h(kbuf + (16 + lq) * 32, hh);
        v8f z = {};
        v8f s0 = wmma_f16(ka0, qf, z);  // S^T tile: (kv_local, q); scale in Q
        v8f s1 = wmma_f16(ka1, qf, z);

        float cm = -3.0e38f;
#pragma unroll
        for (int r = 0; r < 8; ++r) cm = fmaxf(cm, fmaxf(s0[r], s1[r]));
        cm = fmaxf(cm, __shfl_xor(cm, 16, 32));
        const float m_new = fmaxf(m_i, cm);
        const float alpha = __expf(m_i - m_new);

        float p0[8], p1[8], rs = 0.0f;
#pragma unroll
        for (int r = 0; r < 8; ++r) {
            p0[r] = __expf(s0[r] - m_new);
            p1[r] = __expf(s1[r] - m_new);
            rs += p0[r] + p1[r];
        }
        rs += __shfl_xor(rs, 16, 32);
        l_i = l_i * alpha + rs;
        m_i = m_new;
#pragma unroll
        for (int r = 0; r < 8; ++r) { acc0[r] *= alpha; acc1[r] *= alpha; }

        // Build P^T B-fragment: lane needs kv run [16h..16h+15]; it owns
        // {8h..8h+7} of each tile -> swap halves with lane^16.
        v16h pb;
#pragma unroll
        for (int r = 0; r < 8; ++r) {
            const float send = hh ? p0[r] : p1[r];
            const float recv = __shfl_xor(send, 16, 32);
            const float lo = hh ? recv : p0[r];
            const float hi = hh ? p1[r] : recv;
            pb[r] = (_Float16)lo;
            pb[8 + r] = (_Float16)hi;
        }

        // V^T A-fragments (rows d = lq and 16+lq, contiguous kv runs)
        v16h va0 = load_a32_h(vbuf + lq * 32, hh);
        v16h va1 = load_a32_h(vbuf + (16 + lq) * 32, hh);
        acc0 = wmma_f16(va0, pb, acc0);  // ctx^T rows d=0..15
        acc1 = wmma_f16(va1, pb, acc1);  // ctx^T rows d=16..31
    }

    const float inv = 1.0f / l_i;
    v8h o0, o1;
#pragma unroll
    for (int r = 0; r < 8; ++r) {
        o0[r] = (_Float16)(acc0[r] * inv);
        o1[r] = (_Float16)(acc1[r] * inv);
    }
    _Float16* crow =
        Ctx + ((size_t)(b * NSEQ) + qbase + lq) * EMB + head * 32 + 8 * hh;
    *(v8h*)(crow) = o0;
    *(v8h*)(crow + 16) = o1;
}

// ---------------------------------------------------------------------------
// Kernel 3: output projection.  out = Ctx @ Wo^T + bo, f32 result.
// Same out^T = Wo * ctx^T scheme; Ctx is already f16.
// ---------------------------------------------------------------------------
__global__ __launch_bounds__(256) void out_proj_kernel(
    const _Float16* __restrict__ Ctx, const float* __restrict__ Wo,
    const float* __restrict__ bo, float* __restrict__ Out) {
    const int lane = threadIdx.x & 31;
    const int wave = threadIdx.x >> 5;
    const int lq = lane & 15;
    const int hh = lane >> 4;
    const int tok = blockIdx.x * 128 + wave * 16 + lq;

    const _Float16* xr = Ctx + (size_t)tok * EMB;
    v16h xb[4];
#pragma unroll
    for (int c = 0; c < 4; ++c) xb[c] = *(const v16h*)(xr + 32 * c + 16 * hh);

#pragma unroll
    for (int jt = 0; jt < 8; ++jt) {
        const int jbase = jt * 16;
        v8f acc = *(const v8f*)(bo + jbase + 8 * hh);
#pragma unroll
        for (int c = 0; c < 4; ++c) {
            const float* wr = Wo + (size_t)(jbase + lq) * EMB + 32 * c;
            v16h wa = load_a32_f(wr, hh);
            acc = wmma_f16(wa, xb[c], acc);
        }
        *(v8f*)(Out + (size_t)tok * EMB + jbase + 8 * hh) = acc;
    }
}

// ---------------------------------------------------------------------------
extern "C" void kernel_launch(void* const* d_in, const int* in_sizes, int n_in,
                              void* d_out, int out_size, void* d_ws,
                              size_t ws_size, hipStream_t stream) {
    const float* query = (const float*)d_in[0];
    const float* key   = (const float*)d_in[1];
    const float* value = (const float*)d_in[2];
    const float* Wq = (const float*)d_in[3];
    const float* bq = (const float*)d_in[4];
    const float* Wk = (const float*)d_in[5];
    const float* bk = (const float*)d_in[6];
    const float* Wv = (const float*)d_in[7];
    const float* bv = (const float*)d_in[8];
    const float* Wo = (const float*)d_in[9];
    const float* bo = (const float*)d_in[10];
    float* out = (float*)d_out;

    const size_t elems = (size_t)TOKENS * EMB;  // 2,097,152
    _Float16* Qh  = (_Float16*)d_ws;
    _Float16* Kh  = Qh + elems;
    _Float16* Vt  = Kh + elems;
    _Float16* Ctx = Vt + elems;

    dim3 pgrid(TOKENS / 128, 3);
    qkv_proj_kernel<<<pgrid, 256, 0, stream>>>(query, key, value, Wq, Wk, Wv,
                                               bq, bk, bv, Qh, Kh, Vt);

    flash_attn_kernel<<<BATCH * NHEAD * (NSEQ / 128), 256, 0, stream>>>(
        Qh, Kh, Vt, Ctx);

    out_proj_kernel<<<TOKENS / 128, 256, 0, stream>>>(Ctx, Wo, bo, out);
}
